// GAT_26792005992590
// MI455X (gfx1250) — matrix operand order repaired and checked by reading, not compiled, not run on hardware
//
#include <hip/hip_runtime.h>
#include <hip/hip_bf16.h>

#define NUM_LAYERS 4
#define HEADS      3
#define C_MOL      512
#define C_PROT     1280
#define G_MOL      512
#define G_PROT     32
#define NEG_SLOPE  0.2f
#define LN_EPS     1e-5f

typedef __bf16 bf16_t;
typedef __attribute__((ext_vector_type(16))) __bf16 v16bf;
typedef __attribute__((ext_vector_type(8)))  __bf16 v8bf;
typedef __attribute__((ext_vector_type(8)))  float  v8f;

// ---------------------------------------------------------------- helpers
__device__ __forceinline__ bf16_t f2bf(float f) {
  unsigned u = __float_as_uint(f);
  unsigned r = (u + 0x7FFFu + ((u >> 16) & 1u)) >> 16;   // round-to-nearest-even
  unsigned short h = (unsigned short)r;
  bf16_t b; __builtin_memcpy(&b, &h, 2); return b;
}
// monotonic float <-> uint key (for atomicMax-based segment max)
__device__ __forceinline__ unsigned fenc(float x) {
  unsigned u = __float_as_uint(x);
  return (u & 0x80000000u) ? ~u : (u | 0x80000000u);
}
__device__ __forceinline__ float fdec(unsigned k) {
  unsigned u = (k & 0x80000000u) ? (k & 0x7FFFFFFFu) : ~k;
  return __uint_as_float(u);
}
__device__ __forceinline__ void edge_sd(const int* __restrict__ src,
                                        const int* __restrict__ dst,
                                        int E, int e, int& s, int& d) {
  if (e < E) { s = src[e]; d = dst[e]; } else { s = e - E; d = s; }
}

// ---------------------------------------------------------------- conversion
__global__ __launch_bounds__(256) void f32_to_bf16_k(const float* __restrict__ in,
                                                     bf16_t* __restrict__ out, long n) {
  long i = (long)blockIdx.x * blockDim.x + threadIdx.x;
  if (i < n) out[i] = f2bf(in[i]);
}

// W:[K,N] f32 -> Wt:[N,K] bf16 (transposed so B-fragments read contiguous K)
__global__ __launch_bounds__(256) void w_pack_k(const float* __restrict__ W,
                                                bf16_t* __restrict__ Wt, int K, int N) {
  long i = (long)blockIdx.x * blockDim.x + threadIdx.x;
  if (i >= (long)K * N) return;
  int k = (int)(i / N), n = (int)(i % N);
  Wt[(size_t)n * K + k] = f2bf(W[i]);
}

// ---------------------------------------------------------------- WMMA GEMM
// C[M,N] = A[M,K]*B[K,N]; A bf16 row-major, Bt = B^T bf16 [N,K] row-major.
// One wave computes a 32x64 macro-tile (2x4 WMMA tiles, 8 WMMAs per k-step).
// Requires N%64==0 and K%32==0 (true: N in {1536,3840}, K in {512,1280}).
__global__ __launch_bounds__(256) void gemm_bf16_wmma(
    const bf16_t* __restrict__ A, const bf16_t* __restrict__ Bt,
    float* __restrict__ C, int M, int N, int K) {
  const int lane = threadIdx.x & 31;
  const int wave = threadIdx.x >> 5;
  const int gw   = blockIdx.x * 8 + wave;
  const int tilesN = N >> 6;               // 64-wide macro tiles
  const int tm = gw / tilesN;
  const int tn = gw - tm * tilesN;
  if (tm * 32 >= M) return;                // wave-uniform: EXEC stays all-ones

  const int ml = lane & 15;
  const int kh = lane >> 4;

  const bf16_t* ap0 = A + (size_t)(tm * 32 + ml) * K + kh * 16;
  const bf16_t* ap1 = ap0 + (size_t)16 * K;
  const bf16_t* bp[4];
#pragma unroll
  for (int q = 0; q < 4; ++q)
    bp[q] = Bt + (size_t)(tn * 64 + q * 16 + ml) * K + kh * 16;

  v8f acc[2][4];
#pragma unroll
  for (int i = 0; i < 2; ++i)
#pragma unroll
    for (int q = 0; q < 4; ++q) acc[i][q] = (v8f){};

#pragma unroll 2
  for (int k0 = 0; k0 < K; k0 += 32) {
    union { v16bf v; v8bf h[2]; } a0, a1;
    a0.h[0] = *(const v8bf*)(ap0 + k0);
    a0.h[1] = *(const v8bf*)(ap0 + k0 + 8);
    a1.h[0] = *(const v8bf*)(ap1 + k0);
    a1.h[1] = *(const v8bf*)(ap1 + k0 + 8);
    v16bf b[4];
#pragma unroll
    for (int q = 0; q < 4; ++q) b[q] = *(const v16bf*)(bp[q] + k0);
#pragma unroll
    for (int q = 0; q < 4; ++q) {
      acc[0][q] = __builtin_amdgcn_wmma_f32_16x16x32_bf16(false, a0.v, false, b[q], (short)0, acc[0][q], false, false);
      acc[1][q] = __builtin_amdgcn_wmma_f32_16x16x32_bf16(false, a1.v, false, b[q], (short)0, acc[1][q], false, false);
    }
  }

#pragma unroll
  for (int r = 0; r < 8; ++r) {
#pragma unroll
    for (int i = 0; i < 2; ++i) {
      int m = tm * 32 + i * 16 + r + 8 * kh;
      if (m < M) {
        float* cp = C + (size_t)m * N + tn * 64 + ml;
#pragma unroll
        for (int q = 0; q < 4; ++q) cp[q * 16] = acc[i][q][r];
      }
    }
  }
}

// ---------------------------------------------------------------- attention
// one wave per (node, head): al_s = <h[n,hd,:], a_src[hd,:]>, al_d likewise
__global__ __launch_bounds__(32) void attn_logits_k(const float* __restrict__ h,
    const float* __restrict__ asrc, const float* __restrict__ adst,
    float* __restrict__ als, float* __restrict__ ald, int C) {
  int p  = blockIdx.x;
  int n  = p / HEADS, hd = p - n * HEADS;
  const float* hp = h + (size_t)n * HEADS * C + (size_t)hd * C;
  const float* as = asrc + (size_t)hd * C;
  const float* ad = adst + (size_t)hd * C;
  float s0 = 0.f, s1 = 0.f;
  for (int c = threadIdx.x; c < C; c += 32) {
    float v = hp[c]; s0 += v * as[c]; s1 += v * ad[c];
  }
  for (int o = 16; o > 0; o >>= 1) { s0 += __shfl_down(s0, o); s1 += __shfl_down(s1, o); }
  if (threadIdx.x == 0) { als[p] = s0; ald[p] = s1; }
}

__global__ __launch_bounds__(256) void edge_max_k(const int* __restrict__ src,
    const int* __restrict__ dst, int E, int EA,
    const float* __restrict__ als, const float* __restrict__ ald,
    unsigned* __restrict__ mkey) {
  int t = blockIdx.x * blockDim.x + threadIdx.x;
  if (t >= EA * HEADS) return;
  int e = t / HEADS, hd = t - e * HEADS;
  int s, d; edge_sd(src, dst, E, e, s, d);
  float v = als[s * HEADS + hd] + ald[d * HEADS + hd];
  v = v > 0.f ? v : NEG_SLOPE * v;
  atomicMax(&mkey[d * HEADS + hd], fenc(v));
}

__global__ __launch_bounds__(256) void edge_sum_k(const int* __restrict__ src,
    const int* __restrict__ dst, int E, int EA,
    const float* __restrict__ als, const float* __restrict__ ald,
    const unsigned* __restrict__ mkey, float* __restrict__ z) {
  int t = blockIdx.x * blockDim.x + threadIdx.x;
  if (t >= EA * HEADS) return;
  int e = t / HEADS, hd = t - e * HEADS;
  int s, d; edge_sd(src, dst, E, e, s, d);
  float v = als[s * HEADS + hd] + ald[d * HEADS + hd];
  v = v > 0.f ? v : NEG_SLOPE * v;
  atomicAdd(&z[d * HEADS + hd], expf(v - fdec(mkey[d * HEADS + hd])));
}

// per-edge softmax weight: alpha[e,hd]
__global__ __launch_bounds__(256) void alpha_k(const int* __restrict__ src,
    const int* __restrict__ dst, int E, int EA,
    const float* __restrict__ als, const float* __restrict__ ald,
    const unsigned* __restrict__ mkey, const float* __restrict__ z,
    float* __restrict__ alpha) {
  int t = blockIdx.x * blockDim.x + threadIdx.x;
  if (t >= EA * HEADS) return;
  int e = t / HEADS, hd = t - e * HEADS;
  int s, d; edge_sd(src, dst, E, e, s, d);
  float v = als[s * HEADS + hd] + ald[d * HEADS + hd];
  v = v > 0.f ? v : NEG_SLOPE * v;
  float m = fdec(mkey[d * HEADS + hd]);
  alpha[t] = expf(v - m) / (z[d * HEADS + hd] + 1e-16f);
}

// ---------------------------------------------------------------- CSR build (per modality, once)
__global__ __launch_bounds__(256) void deg_k(const int* __restrict__ src,
    const int* __restrict__ dst, int E, int EA, int* __restrict__ deg) {
  int e = blockIdx.x * blockDim.x + threadIdx.x;
  if (e >= EA) return;
  int s, d; edge_sd(src, dst, E, e, s, d);
  atomicAdd(&deg[d], 1);
}

// single-block chunked exclusive scan: off[0..n], off[n] = total
__global__ __launch_bounds__(1024) void scan_k(const int* __restrict__ deg,
                                               int* __restrict__ off, int n) {
  __shared__ int sm[1024];
  __shared__ int carry;
  if (threadIdx.x == 0) carry = 0;
  __syncthreads();
  for (int base = 0; base < n; base += 1024) {
    int i = base + (int)threadIdx.x;
    int v = (i < n) ? deg[i] : 0;
    sm[threadIdx.x] = v;
    __syncthreads();
    for (int o = 1; o < 1024; o <<= 1) {
      int t = (threadIdx.x >= (unsigned)o) ? sm[threadIdx.x - o] : 0;
      __syncthreads();
      sm[threadIdx.x] += t;
      __syncthreads();
    }
    if (i < n) off[i] = carry + sm[threadIdx.x] - v;   // exclusive
    __syncthreads();
    if (threadIdx.x == 1023) carry += sm[1023];
    __syncthreads();
  }
  if (threadIdx.x == 0) off[n] = carry;
}

__global__ __launch_bounds__(256) void csr_fill_k(const int* __restrict__ src,
    const int* __restrict__ dst, int E, int EA,
    int* __restrict__ cursor, int* __restrict__ eidx) {
  int e = blockIdx.x * blockDim.x + threadIdx.x;
  if (e >= EA) return;
  int s, d; edge_sd(src, dst, E, e, s, d);
  int pos = atomicAdd(&cursor[d], 1);
  eidx[pos] = e;
}

// ---------------------------------------------------------------- gather aggregation
// block per dst node: acc[d,hd,c] = sum_{e->d} alpha[e,hd]*h[src(e),hd,c]
// HC must be a multiple of 256 (1536, 3840). No atomics, no pre-zeroing.
__global__ __launch_bounds__(256) void agg_gather_k(const float* __restrict__ h,
    const int* __restrict__ src, const int* __restrict__ dst, int E,
    const int* __restrict__ off, const int* __restrict__ eidx,
    const float* __restrict__ alpha, float* __restrict__ acc, int C) {
  const int d   = blockIdx.x;
  const int beg = off[d], end = off[d + 1];
  const int HC  = HEADS * C;
  __shared__ int   s_s[64];
  __shared__ float s_a[64 * HEADS];

  float accv[15];
#pragma unroll
  for (int i = 0; i < 15; ++i) accv[i] = 0.f;

  int hds[15];
  {
    int i = 0;
    for (int c = threadIdx.x; c < HC; c += 256, ++i) hds[i] = c / C;
  }

  for (int base = beg; base < end; base += 64) {
    int cnt = min(64, end - base);
    __syncthreads();
    if ((int)threadIdx.x < cnt) {
      int e = eidx[base + threadIdx.x];
      int s, dd; edge_sd(src, dst, E, e, s, dd);
      s_s[threadIdx.x] = s;
#pragma unroll
      for (int hh = 0; hh < HEADS; ++hh)
        s_a[threadIdx.x * HEADS + hh] = alpha[(size_t)e * HEADS + hh];
    }
    __syncthreads();
    for (int j = 0; j < cnt; ++j) {
      const float* hs = h + (size_t)s_s[j] * HC;
      const float* aj = s_a + j * HEADS;
      int i = 0;
      for (int c = threadIdx.x; c < HC; c += 256, ++i)
        accv[i] += aj[hds[i]] * hs[c];
    }
  }
  float* ap = acc + (size_t)d * HC;
  {
    int i = 0;
    for (int c = threadIdx.x; c < HC; c += 256, ++i) ap[c] = accv[i];
  }
}

// ---------------------------------------------------------------- head-mean + bias + LN
__global__ __launch_bounds__(256) void headmean_ln_k(const float* __restrict__ acc,
    const float* __restrict__ bias, const float* __restrict__ gam,
    const float* __restrict__ bet, float* __restrict__ xout, int C) {
  int n = blockIdx.x;
  __shared__ float ybuf[C_PROT];
  __shared__ float rsum[256], rsq[256];
  const float* a = acc + (size_t)n * HEADS * C;
  float ls = 0.f, lq = 0.f;
  for (int c = threadIdx.x; c < C; c += 256) {
    float y = 0.f;
#pragma unroll
    for (int hh = 0; hh < HEADS; ++hh) y += a[(size_t)hh * C + c];
    y = y * (1.0f / HEADS) + bias[c];
    ybuf[c] = y; ls += y; lq += y * y;
  }
  rsum[threadIdx.x] = ls; rsq[threadIdx.x] = lq;
  __syncthreads();
  for (int o = 128; o > 0; o >>= 1) {
    if (threadIdx.x < o) { rsum[threadIdx.x] += rsum[threadIdx.x + o]; rsq[threadIdx.x] += rsq[threadIdx.x + o]; }
    __syncthreads();
  }
  float mu  = rsum[0] / C;
  float var = rsq[0] / C - mu * mu;
  float inv = rsqrtf(var + LN_EPS);
  for (int c = threadIdx.x; c < C; c += 256)
    xout[(size_t)n * C + c] = (ybuf[c] - mu) * inv * gam[c] + bet[c];
}

// ---------------------------------------------------------------- pooling & embedding
__global__ __launch_bounds__(256) void count_k(const int* __restrict__ batch,
                                               float* __restrict__ cnt, int Nn) {
  int i = blockIdx.x * blockDim.x + threadIdx.x;
  if (i < Nn) atomicAdd(&cnt[batch[i]], 1.0f);
}
__global__ __launch_bounds__(256) void pool_add_k(const float* __restrict__ x,
    const int* __restrict__ batch, float* __restrict__ pooled, int C) {
  int n = blockIdx.x, g = batch[n];
  for (int c = threadIdx.x; c < C; c += 256)
    atomicAdd(&pooled[(size_t)g * C + c], x[(size_t)n * C + c]);
}
__global__ __launch_bounds__(256) void pool_fin_k(const float* __restrict__ pooled,
    const float* __restrict__ cnt, float* __restrict__ out, int C, int li) {
  int g = blockIdx.x;
  float inv = 1.0f / fmaxf(cnt[g], 1.0f);
  for (int c = threadIdx.x; c < C; c += 256)
    out[(size_t)g * (NUM_LAYERS + 1) * C + (size_t)li * C + c] = pooled[(size_t)g * C + c] * inv;
}
__global__ __launch_bounds__(256) void emb_mean_k(const int* __restrict__ tok,
    const float* __restrict__ emb, float* __restrict__ x, int C) {
  int n = blockIdx.x;
  int r[9];
#pragma unroll
  for (int j = 0; j < 9; ++j) r[j] = tok[n * 9 + j];
  for (int c = threadIdx.x; c < C; c += 256) {
    float s = 0.f;
#pragma unroll
    for (int j = 0; j < 9; ++j) s += emb[(size_t)r[j] * C + c];
    x[(size_t)n * C + c] = s * (1.0f / 9.0f);
  }
}

// ---------------------------------------------------------------- host
static inline char* ws_take(char*& p, size_t bytes) {
  char* r = p; p += (bytes + 255) & ~(size_t)255; return r;
}
static inline int idiv(long a, long b) { return (int)((a + b - 1) / b); }

extern "C" void kernel_launch(void* const* d_in, const int* in_sizes, int n_in,
                              void* d_out, int out_size, void* d_ws, size_t ws_size,
                              hipStream_t stream) {
  (void)n_in; (void)out_size; (void)ws_size;
  const int*   drug_x   = (const int*)d_in[0];
  const int*   mol_ei   = (const int*)d_in[1];
  const int*   mol_bat  = (const int*)d_in[2];
  const float* prot_x   = (const float*)d_in[3];
  const int*   prot_ei  = (const int*)d_in[4];
  const int*   prot_bat = (const int*)d_in[5];
  const float* emb      = (const float*)d_in[6];
  const float* mol_W    = (const float*)d_in[7];
  const float* mol_as   = (const float*)d_in[8];
  const float* mol_ad   = (const float*)d_in[9];
  const float* mol_b    = (const float*)d_in[10];
  const float* mol_g    = (const float*)d_in[11];
  const float* mol_bb   = (const float*)d_in[12];
  const float* prot_W   = (const float*)d_in[13];
  const float* prot_as  = (const float*)d_in[14];
  const float* prot_ad  = (const float*)d_in[15];
  const float* prot_b   = (const float*)d_in[16];
  const float* prot_g   = (const float*)d_in[17];
  const float* prot_bb  = (const float*)d_in[18];
  float* out = (float*)d_out;

  const int n_mol  = in_sizes[0] / 9;
  const int e_mol  = in_sizes[1] / 2;
  const int n_prot = in_sizes[3] / C_PROT;
  const int e_prot = in_sizes[4] / 2;

  const size_t mpad_mol  = ((size_t)n_mol  + 31) & ~(size_t)31;
  const size_t mpad_prot = ((size_t)n_prot + 31) & ~(size_t)31;
  const size_t npadmax = (mpad_mol > mpad_prot) ? mpad_mol : mpad_prot;
  const size_t x_el  = (mpad_mol * C_MOL > mpad_prot * C_PROT) ? mpad_mol * C_MOL : mpad_prot * C_PROT;
  const size_t h_el  = x_el * HEADS;
  const size_t w_el  = ((size_t)C_PROT * HEADS * C_PROT);   // >= mol weight size
  const size_t nh_el = npadmax * HEADS;
  const size_t ea_max = (size_t)((e_mol + n_mol > e_prot + n_prot) ? e_mol + n_mol : e_prot + n_prot);
  const size_t po_el = ((size_t)G_MOL * C_MOL > (size_t)G_PROT * C_PROT)
                           ? (size_t)G_MOL * C_MOL : (size_t)G_PROT * C_PROT;

  char* p = (char*)d_ws;
  float*    xbuf   = (float*)   ws_take(p, x_el * 4);
  bf16_t*   xb16   = (bf16_t*)  ws_take(p, x_el * 2);
  float*    hbuf   = (float*)   ws_take(p, h_el * 4);
  float*    accbuf = (float*)   ws_take(p, h_el * 4);
  bf16_t*   wtb16  = (bf16_t*)  ws_take(p, w_el * 2);
  float*    als    = (float*)   ws_take(p, nh_el * 4);
  float*    ald    = (float*)   ws_take(p, nh_el * 4);
  unsigned* mkey   = (unsigned*)ws_take(p, nh_el * 4);
  float*    zbuf   = (float*)   ws_take(p, nh_el * 4);
  float*    alpha  = (float*)   ws_take(p, ea_max * HEADS * 4);
  int*      deg    = (int*)     ws_take(p, npadmax * 4);
  int*      off    = (int*)     ws_take(p, (npadmax + 1) * 4);
  int*      cursor = (int*)     ws_take(p, npadmax * 4);
  int*      eidx   = (int*)     ws_take(p, ea_max * 4);
  float*    pooled = (float*)   ws_take(p, po_el * 4);
  float*    cnt    = (float*)   ws_take(p, (size_t)G_MOL * 4);

  auto run_modality = [&](int Nn, int E, int C, int G,
                          const int* ei, const int* batch,
                          const float* W, const float* Asrc, const float* Adst,
                          const float* bias, const float* lng, const float* lnb,
                          float* outbase) {
    const int Hc = HEADS * C;
    const int* src = ei;
    const int* dst = ei + E;
    const int EA = E + Nn;

    // graph-constant prep: counts + dst-CSR
    hipMemsetAsync(cnt, 0, (size_t)G * 4, stream);
    count_k<<<idiv(Nn, 256), 256, 0, stream>>>(batch, cnt, Nn);
    hipMemsetAsync(deg, 0, (size_t)Nn * 4, stream);
    deg_k<<<idiv(EA, 256), 256, 0, stream>>>(src, dst, E, EA, deg);
    scan_k<<<1, 1024, 0, stream>>>(deg, off, Nn);
    hipMemcpyAsync(cursor, off, (size_t)Nn * 4, hipMemcpyDeviceToDevice, stream);
    csr_fill_k<<<idiv(EA, 256), 256, 0, stream>>>(src, dst, E, EA, cursor, eidx);

    // feats[0]
    hipMemsetAsync(pooled, 0, (size_t)G * C * 4, stream);
    pool_add_k<<<Nn, 256, 0, stream>>>(xbuf, batch, pooled, C);
    pool_fin_k<<<G, 256, 0, stream>>>(pooled, cnt, outbase, C, 0);

    for (int li = 0; li < NUM_LAYERS; ++li) {
      const float* Wl = W + (size_t)li * C * Hc;
      long wn = (long)C * Hc;
      w_pack_k<<<idiv(wn, 256), 256, 0, stream>>>(Wl, wtb16, C, Hc);
      long xn = (long)Nn * C;
      f32_to_bf16_k<<<idiv(xn, 256), 256, 0, stream>>>(xbuf, xb16, xn);

      int tiles = idiv(Nn, 32) * (Hc >> 6);
      gemm_bf16_wmma<<<idiv(tiles, 8), 256, 0, stream>>>(xb16, wtb16, hbuf, Nn, Hc, C);

      attn_logits_k<<<Nn * HEADS, 32, 0, stream>>>(hbuf, Asrc + (size_t)li * HEADS * C,
                                                   Adst + (size_t)li * HEADS * C, als, ald, C);

      hipMemsetAsync(mkey, 0, (size_t)Nn * HEADS * 4, stream);
      hipMemsetAsync(zbuf, 0, (size_t)Nn * HEADS * 4, stream);
      int tmax = EA * HEADS;
      edge_max_k<<<idiv(tmax, 256), 256, 0, stream>>>(src, dst, E, EA, als, ald, mkey);
      edge_sum_k<<<idiv(tmax, 256), 256, 0, stream>>>(src, dst, E, EA, als, ald, mkey, zbuf);
      alpha_k<<<idiv(tmax, 256), 256, 0, stream>>>(src, dst, E, EA, als, ald, mkey, zbuf, alpha);

      agg_gather_k<<<Nn, 256, 0, stream>>>(hbuf, src, dst, E, off, eidx, alpha, accbuf, C);

      headmean_ln_k<<<Nn, 256, 0, stream>>>(accbuf, bias + (size_t)li * C,
                                            lng + (size_t)li * C, lnb + (size_t)li * C, xbuf, C);

      hipMemsetAsync(pooled, 0, (size_t)G * C * 4, stream);
      pool_add_k<<<Nn, 256, 0, stream>>>(xbuf, batch, pooled, C);
      pool_fin_k<<<G, 256, 0, stream>>>(pooled, cnt, outbase, C, li + 1);
    }
  };

  // ---- molecule branch ----
  emb_mean_k<<<n_mol, 256, 0, stream>>>(drug_x, emb, xbuf, C_MOL);
  run_modality(n_mol, e_mol, C_MOL, G_MOL, mol_ei, mol_bat,
               mol_W, mol_as, mol_ad, mol_b, mol_g, mol_bb, out);

  // ---- protein branch (reuses the same ws buffers) ----
  hipMemcpyAsync(xbuf, prot_x, (size_t)n_prot * C_PROT * 4, hipMemcpyDeviceToDevice, stream);
  run_modality(n_prot, e_prot, C_PROT, G_PROT, prot_ei, prot_bat,
               prot_W, prot_as, prot_ad, prot_b, prot_g, prot_bb,
               out + (size_t)G_MOL * (NUM_LAYERS + 1) * C_MOL);
}
